// TransE_65180423684121
// MI455X (gfx1250) — matrix operand order repaired and checked by reading, not compiled, not run on hardware
//
#include <hip/hip_runtime.h>
#include <hip/hip_bf16.h>

typedef __attribute__((ext_vector_type(16))) __bf16        v16bf;
typedef __attribute__((ext_vector_type(8)))  float         v8f;
typedef __attribute__((ext_vector_type(4)))  unsigned int  u32x4;

#define BATCH 65536
#define D_OUT 200     // D
#define D_EXT 300     // DEXT
#define KPAD  320     // DEXT padded to multiple of 32
#define NPAD  208     // D padded to multiple of 16
#define NT    13      // N tiles (208/16)
#define KT    10      // K steps (320/32)

union Frag16 {
    v16bf          v;
    unsigned short u[16];
    u32x4          q[2];
};

__device__ __forceinline__ unsigned short f2bf(float f) {
    unsigned int u = __float_as_uint(f);
    unsigned int r = u + 0x7FFFu + ((u >> 16) & 1u);   // round-to-nearest-even
    return (unsigned short)(r >> 16);
}

// Convert W [200,300] f32 (row-major, [out,in]) into padded bf16 tile buffer
// wt[n][k], n in [0,208), k in [0,320), zero-padded. Column (n)-major with k
// contiguous => each WMMA B-fragment lane reads 2 runs of 8 contiguous bf16.
__global__ void prep_w_bf16(const float* __restrict__ W,
                            unsigned short* __restrict__ wt) {
    int i = blockIdx.x * blockDim.x + threadIdx.x;
    if (i >= NPAD * KPAD) return;
    int n = i / KPAD;
    int k = i - n * KPAD;
    float v = (n < D_OUT && k < D_EXT) ? W[n * D_EXT + k] : 0.0f;
    wt[i] = f2bf(v);
}

__global__ __launch_bounds__(256)
void transe_fused_kernel(const int* __restrict__ h_p, const int* __restrict__ t_p,
                         const int* __restrict__ r_p, const int* __restrict__ h_n,
                         const int* __restrict__ t_n, const int* __restrict__ r_n,
                         const float* __restrict__ entity,
                         const float* __restrict__ relation,
                         const float* __restrict__ ext,
                         const unsigned short* __restrict__ wt,
                         const float* __restrict__ bias,
                         const float* __restrict__ w_in_p,
                         const float* __restrict__ w_ext_p,
                         float* __restrict__ out) {
    __shared__ unsigned short A_lds[32 * KPAD];   // 16 h rows + 16 t rows, bf16
    __shared__ float trans_lds[32 * NPAD];        // relu(X @ Wt + b), f32
    __shared__ int hidx[16], tidx[16], ridx[16];

    const int tid  = threadIdx.x;
    const int base = blockIdx.x * 16;
    const int neg  = blockIdx.y;                  // 0 = positive, 1 = negative

    const int* hsel = neg ? h_n : h_p;
    const int* tsel = neg ? t_n : t_p;
    const int* rsel = neg ? r_n : r_p;

    if (tid < 16)       hidx[tid]      = hsel[base + tid];
    else if (tid < 32)  tidx[tid - 16] = tsel[base + tid - 16];
    else if (tid < 48)  ridx[tid - 32] = rsel[base + tid - 32];
    __syncthreads();

    // ---- Stage 1: gather ext_emb rows -> LDS as bf16, K zero-padded ----
    for (int i = tid; i < 32 * KPAD; i += 256) {
        int row = i / KPAD;
        int k   = i - row * KPAD;
        int idx = (row < 16) ? hidx[row] : tidx[row - 16];
        float v = (k < D_EXT) ? ext[(size_t)idx * D_EXT + k] : 0.0f;
        A_lds[i] = f2bf(v);
    }
    __syncthreads();

    // ---- Stage 2: WMMA GEMMs: trans = relu(A @ Wt + b) for h and t ----
    const int wave = tid >> 5;
    const int lane = tid & 31;
    const int half = lane >> 4;   // which half-wave
    const int l16  = lane & 15;

    for (int t = wave; t < 2 * NT; t += 8) {
        const int mat = t / NT;   // 0 = h rows, 1 = t rows
        const int nt  = t % NT;
        v8f acc = {0.f, 0.f, 0.f, 0.f, 0.f, 0.f, 0.f, 0.f};

        // A fragment: lane (m=l16), halves hold K {0..7,16..23} / {8..15,24..31}
        const unsigned short* arow = &A_lds[(mat * 16 + l16) * KPAD + half * 8];
        // B fragment: lane holds column n = nt*16+l16, halves hold K {0..7,16..23} / {8..15,24..31}
        const unsigned short* bcol = &wt[(nt * 16 + l16) * KPAD + half * 8];

#pragma unroll
        for (int kt = 0; kt < KT; ++kt) {
            Frag16 a, bb;
            const unsigned short* ap = arow + kt * 32;
            a.q[0] = *(const u32x4*)(ap);
            a.q[1] = *(const u32x4*)(ap + 16);
            const unsigned short* bp = bcol + kt * 32;
            bb.q[0] = *(const u32x4*)(bp);
            bb.q[1] = *(const u32x4*)(bp + 16);
            acc = __builtin_amdgcn_wmma_f32_16x16x32_bf16(
                false, a.v, false, bb.v, (short)0, acc, false, false);
        }

        const int n = nt * 16 + l16;
        const float bv = (n < D_OUT) ? bias[n] : 0.0f;
#pragma unroll
        for (int v = 0; v < 8; ++v) {
            const int m = v + half * 8;                  // output row within tile
            float x = acc[v] + bv;
            trans_lds[(mat * 16 + m) * NPAD + n] = x > 0.0f ? x : 0.0f;
        }
    }
    __syncthreads();

    // ---- Stage 3: combine + distance: 16 threads per batch row ----
    const float w_in  = w_in_p[0];
    const float w_ext = w_ext_p[0];
    const int m    = tid >> 4;     // row 0..15
    const int tcol = tid & 15;

    const float* eh = entity   + (size_t)hidx[m] * D_OUT;
    const float* et = entity   + (size_t)tidx[m] * D_OUT;
    const float* rl = relation + (size_t)ridx[m] * D_OUT;

    float s = 0.0f;
    for (int n = tcol; n < D_OUT; n += 16) {
        float d = w_in * (eh[n] - et[n]) + rl[n]
                + w_ext * (trans_lds[m * NPAD + n] - trans_lds[(16 + m) * NPAD + n]);
        s += d * d;
    }
#pragma unroll
    for (int off = 8; off > 0; off >>= 1) s += __shfl_xor(s, off, 16);
    if (tcol == 0)
        out[(size_t)neg * BATCH + base + m] = sqrtf(s);
}

extern "C" void kernel_launch(void* const* d_in, const int* in_sizes, int n_in,
                              void* d_out, int out_size, void* d_ws, size_t ws_size,
                              hipStream_t stream) {
    const int*   h_p      = (const int*)  d_in[0];
    const int*   t_p      = (const int*)  d_in[1];
    const int*   r_p      = (const int*)  d_in[2];
    const int*   h_n      = (const int*)  d_in[3];
    const int*   t_n      = (const int*)  d_in[4];
    const int*   r_n      = (const int*)  d_in[5];
    const float* entity   = (const float*)d_in[6];
    const float* relation = (const float*)d_in[7];
    const float* ext      = (const float*)d_in[8];
    const float* W        = (const float*)d_in[9];
    const float* bias     = (const float*)d_in[10];
    const float* w_in     = (const float*)d_in[11];
    const float* w_ext    = (const float*)d_in[12];
    float*       out      = (float*)d_out;

    unsigned short* wt = (unsigned short*)d_ws;   // 208*320*2 = 133,120 bytes

    // Stage 0: convert/pad W -> bf16 B-tile layout (L2-resident afterwards)
    {
        int n = NPAD * KPAD;
        prep_w_bf16<<<(n + 255) / 256, 256, 0, stream>>>(W, wt);
    }

    // Fused gather + WMMA GEMM + combine + L2 distance.
    dim3 grid(BATCH / 16, 2, 1);
    transe_fused_kernel<<<grid, 256, 0, stream>>>(
        h_p, t_p, r_p, h_n, t_n, r_n,
        entity, relation, ext, wt, bias, w_in, w_ext, out);
}
